// GCN_35545149341945
// MI455X (gfx1250) — matrix operand order repaired and checked by reading, not compiled, not run on hardware
//
#include <hip/hip_runtime.h>

typedef __bf16 bf16;
typedef bf16  v16bf __attribute__((ext_vector_type(16)));
typedef bf16  v8bf  __attribute__((ext_vector_type(8)));
typedef bf16  v4bf  __attribute__((ext_vector_type(4)));
typedef float v8f   __attribute__((ext_vector_type(8)));
typedef float v4f   __attribute__((ext_vector_type(4)));

#define CAT16(lo, hi) __builtin_shufflevector((lo), (hi), 0,1,2,3,4,5,6,7,8,9,10,11,12,13,14,15)

// Problem constants (B=4, N=512, E=64, D=256)
#define BN_TOTAL   2048
#define E_         64
#define D_         256
#define K2         512     // 2*D
#define CAT_PITCH  520     // bf16 elems per row; 260 dwords -> lane stride 4 banks
#define INFO_PITCH 260     // f32 elems per row

// ---------------------------------------------------------------------------
// Kernel 0: one-shot f32 -> bf16 weight conversion into scratch
// ---------------------------------------------------------------------------
__global__ __launch_bounds__(256)
void convert_weights(const float* __restrict__ Wgate,
                     const float* __restrict__ Winfo,
                     const float* __restrict__ Wgraph,
                     bf16* __restrict__ wsGate,
                     bf16* __restrict__ wsInfo,
                     bf16* __restrict__ wsGraph) {
    int i = blockIdx.x * 256 + threadIdx.x;
    if (i < 131072) {                 // 256*512
        wsGate[i] = (bf16)Wgate[i];
        wsInfo[i] = (bf16)Winfo[i];
    }
    if (i < 65536) {                  // 256*256
        wsGraph[i] = (bf16)Wgraph[i];
    }
}

// ---------------------------------------------------------------------------
// Kernel 1: per-(b,n) fused gate/info GEMM + attention + weighted reduce
//   grid = 2048 blocks of 256 threads (8 waves)
// ---------------------------------------------------------------------------
__global__ __launch_bounds__(256)
void fused_gat_kernel(const float* __restrict__ stEmbed,
                      const float* __restrict__ edEmbed,
                      const float* __restrict__ edgeEmbed,
                      const unsigned char* __restrict__ mask,
                      const bf16* __restrict__ WgateB,
                      const float* __restrict__ bgate,
                      const bf16* __restrict__ WinfoB,
                      const float* __restrict__ binfo,
                      const float* __restrict__ Wa,
                      const float* __restrict__ ba,
                      const float* __restrict__ Wh,
                      const float* __restrict__ bh,
                      bf16* __restrict__ aOut) {
    __shared__ bf16  catLDS[E_ * CAT_PITCH];    // 66560 B, cat = [edge | ed]
    __shared__ float infoLDS[E_ * INFO_PITCH];  // 66560 B
    __shared__ float WaLDS[D_];
    __shared__ float aStLDS[E_];
    __shared__ float scoreLDS[E_];
    __shared__ float wLDS[E_];
    __shared__ float redLDS[256];

    const int tid = threadIdx.x;
    const int bn  = blockIdx.x;
    const long base = (long)bn * (E_ * D_);

    // ---- Phase 1: stage Wa, build bf16 cat matrix in LDS -------------------
    WaLDS[tid] = Wa[tid];   // D_ == blockDim == 256

#pragma unroll 4
    for (int i = 0; i < 16; ++i) {
        int idx4 = tid + i * 256;           // 0..4095 float4s per source
        int e    = idx4 >> 6;               // row (edge)
        int d4   = (idx4 & 63) << 2;        // col within D
        // Streamed exactly once -> non-temporal, keep L2 for the weights.
        v4f g = __builtin_nontemporal_load((const v4f*)(edgeEmbed + base + e * D_ + d4));
        v4f s = __builtin_nontemporal_load((const v4f*)(edEmbed   + base + e * D_ + d4));
        bf16* rowp = catLDS + e * CAT_PITCH;
        v4bf gb, sb;
        gb[0] = (bf16)g[0]; gb[1] = (bf16)g[1]; gb[2] = (bf16)g[2]; gb[3] = (bf16)g[3];
        sb[0] = (bf16)s[0]; sb[1] = (bf16)s[1]; sb[2] = (bf16)s[2]; sb[3] = (bf16)s[3];
        *(v4bf*)(rowp + d4)       = gb;     // cols 0..255   = edgeEmbed
        *(v4bf*)(rowp + D_ + d4)  = sb;     // cols 256..511 = edEmbed
    }
    __syncthreads();

    // ---- Phase 2: a_st[e] = dot(stEmbed[e,:], Wa) + ba ---------------------
    {
        int e = tid >> 2, q = tid & 3;
        const float* sp = stEmbed + base + e * D_ + q * 64;
        float s = 0.f;
#pragma unroll 4
        for (int j = 0; j < 16; ++j) {
            v4f c = __builtin_nontemporal_load((const v4f*)(sp + j * 4));
            s += c[0] * WaLDS[q * 64 + j * 4 + 0];
            s += c[1] * WaLDS[q * 64 + j * 4 + 1];
            s += c[2] * WaLDS[q * 64 + j * 4 + 2];
            s += c[3] * WaLDS[q * 64 + j * 4 + 3];
        }
        redLDS[tid] = s;
    }
    __syncthreads();
    if (tid < E_) {
        aStLDS[tid] = redLDS[tid * 4] + redLDS[tid * 4 + 1] +
                      redLDS[tid * 4 + 2] + redLDS[tid * 4 + 3] + ba[0];
    }

    // ---- Phase 3: WMMA GEMM  [64 x 512] x [512 x 256] for gate & info ------
    const int w  = tid >> 5;          // wave 0..7
    const int l  = tid & 31;
    const int lm = l & 15;
    const int lh = l >> 4;

#pragma unroll 1
    for (int p = 0; p < 8; ++p) {
        int pr = w * 8 + p;
        int rt = pr & 3;              // row tile of E (4 x 16)
        int ct = pr >> 2;             // col tile of D (16 x 16)
        v8f accg = {}; v8f acci = {};

        // A k order used here (B matches): lane lm holds row rt*16+lm; 16 contiguous K per half
        // offset by 16 for upper lane half.
        const bf16* arow  = catLDS + (rt * 16 + lm) * CAT_PITCH + lh * 16;
        // B layout: lane lm = output col d = ct*16+lm; 16 contiguous K per half.
        const bf16* bgrow = WgateB + (ct * 16 + lm) * K2 + lh * 16;
        const bf16* birow = WinfoB + (ct * 16 + lm) * K2 + lh * 16;

#pragma unroll 4
        for (int k0 = 0; k0 < K2; k0 += 32) {
            v8bf a0 = *(const v8bf*)(arow + k0);
            v8bf a1 = *(const v8bf*)(arow + k0 + 8);
            v16bf af = CAT16(a0, a1);
            v8bf g0 = *(const v8bf*)(bgrow + k0);
            v8bf g1 = *(const v8bf*)(bgrow + k0 + 8);
            v16bf gf = CAT16(g0, g1);
            v8bf i0 = *(const v8bf*)(birow + k0);
            v8bf i1 = *(const v8bf*)(birow + k0 + 8);
            v16bf jf = CAT16(i0, i1);
            accg = __builtin_amdgcn_wmma_f32_16x16x32_bf16(
                false, af, false, gf, (short)0, accg, false, false);
            acci = __builtin_amdgcn_wmma_f32_16x16x32_bf16(
                false, af, false, jf, (short)0, acci, false, false);
        }

        // Epilogue: info = (lin_info + binfo) * sigmoid(lin_gate + bgate)
        float bg = bgate[ct * 16 + lm];
        float bi = binfo[ct * 16 + lm];
        float* op = infoLDS + (rt * 16 + 8 * lh) * INFO_PITCH + ct * 16 + lm;
#pragma unroll
        for (int r = 0; r < 8; ++r) {
            float gate = 1.f / (1.f + __expf(-(accg[r] + bg)));
            op[r * INFO_PITCH] = (acci[r] + bi) * gate;
        }
    }
    __syncthreads();

    // ---- Phase 4: attention score + masked softmax over E ------------------
    if (tid < E_) {
        const float* ip = infoLDS + tid * INFO_PITCH;
        float s = 0.f;
#pragma unroll 8
        for (int d = 0; d < D_; ++d) s += ip[d] * WaLDS[d];
        float a_in = s + ba[0];
        float h  = aStLDS[tid] * Wh[0] + a_in * Wh[1] + bh[0];
        float sc = h > 0.f ? h : 0.01f * h;          // leaky_relu(0.01)
        if (mask[(long)bn * E_ + tid]) sc = -INFINITY;
        scoreLDS[tid] = sc;
    }
    __syncthreads();

    float mx = -INFINITY;
#pragma unroll 8
    for (int e = 0; e < E_; ++e) mx = fmaxf(mx, scoreLDS[e]);
    float sum = 0.f;
#pragma unroll 8
    for (int e = 0; e < E_; ++e) sum += __expf(scoreLDS[e] - mx);
    if (tid < E_) wLDS[tid] = __expf(scoreLDS[tid] - mx) / sum;
    __syncthreads();

    // ---- Phase 5: a[d] = sum_e info[e,d] * w[e]; scale 1/sqrt(D); bf16 out -
    {
        float s = 0.f;
#pragma unroll 8
        for (int e = 0; e < E_; ++e) s += infoLDS[e * INFO_PITCH + tid] * wLDS[e];
        aOut[(long)bn * D_ + tid] = (bf16)(s * 0.0625f);   // 1/sqrt(256)
    }
}

// ---------------------------------------------------------------------------
// Kernel 2: out = relu(aScaled @ Wgraph.T + bgraph)  — [2048x256]x[256x256]
//   grid = 256 blocks x 256 threads (8 waves); one 16x16 tile per wave
// ---------------------------------------------------------------------------
__global__ __launch_bounds__(256)
void out_proj_kernel(const bf16* __restrict__ A,       // [2048 x 256] bf16
                     const bf16* __restrict__ Wg,      // [256 x 256]  bf16
                     const float* __restrict__ bgraph,
                     float* __restrict__ out) {
    const int tid = threadIdx.x;
    const int w = tid >> 5, l = tid & 31, lm = l & 15, lh = l >> 4;
    const int gw = blockIdx.x * 8 + w;     // 0..2047 tiles
    const int mt = gw >> 4;                // 0..127 row tile
    const int ct = gw & 15;                // 0..15  col tile

    v8f acc = {};
    const bf16* ap = A  + (mt * 16 + lm) * D_ + lh * 16;
    const bf16* bp = Wg + (ct * 16 + lm) * D_ + lh * 16;
#pragma unroll 4
    for (int k0 = 0; k0 < D_; k0 += 32) {
        v8bf a0 = *(const v8bf*)(ap + k0);
        v8bf a1 = *(const v8bf*)(ap + k0 + 8);
        v8bf b0 = *(const v8bf*)(bp + k0);
        v8bf b1 = *(const v8bf*)(bp + k0 + 8);
        v16bf af = CAT16(a0, a1);
        v16bf bf_ = CAT16(b0, b1);
        acc = __builtin_amdgcn_wmma_f32_16x16x32_bf16(
            false, af, false, bf_, (short)0, acc, false, false);
    }
    float bias = bgraph[ct * 16 + lm];
#pragma unroll
    for (int r = 0; r < 8; ++r) {
        int m = mt * 16 + r + 8 * lh;
        float v = acc[r] + bias;
        // Write-once output -> non-temporal store.
        __builtin_nontemporal_store(v > 0.f ? v : 0.f,
                                    out + (long)m * D_ + ct * 16 + lm);
    }
}

// ---------------------------------------------------------------------------
extern "C" void kernel_launch(void* const* d_in, const int* in_sizes, int n_in,
                              void* d_out, int out_size, void* d_ws, size_t ws_size,
                              hipStream_t stream) {
    const float*         stEmbed   = (const float*)d_in[0];
    const float*         edEmbed   = (const float*)d_in[1];
    const float*         edgeEmbed = (const float*)d_in[2];
    const unsigned char* mask      = (const unsigned char*)d_in[3];
    const float*         Wgraph    = (const float*)d_in[4];
    const float*         bgraph    = (const float*)d_in[5];
    const float*         Wgate     = (const float*)d_in[6];
    const float*         bgate     = (const float*)d_in[7];
    const float*         Winfo     = (const float*)d_in[8];
    const float*         binfo     = (const float*)d_in[9];
    const float*         Wa        = (const float*)d_in[10];
    const float*         ba        = (const float*)d_in[11];
    const float*         Wh        = (const float*)d_in[12];
    const float*         bh       = (const float*)d_in[13];

    // Scratch layout (bf16): Wgate[131072] | Winfo[131072] | Wgraph[65536] | a[524288]
    bf16* wsGate  = (bf16*)d_ws;
    bf16* wsInfo  = wsGate  + 131072;
    bf16* wsGraph = wsInfo  + 131072;
    bf16* wsA     = wsGraph + 65536;

    convert_weights<<<512, 256, 0, stream>>>(Wgate, Winfo, Wgraph,
                                             wsGate, wsInfo, wsGraph);
    fused_gat_kernel<<<BN_TOTAL, 256, 0, stream>>>(
        stEmbed, edEmbed, edgeEmbed, mask,
        wsGate, bgate, wsInfo, binfo, Wa, ba, Wh, bh, wsA);
    out_proj_kernel<<<256, 256, 0, stream>>>(wsA, wsGraph, bgraph, (float*)d_out);
}
